// CasualSelfAttention_4260607558134
// MI455X (gfx1250) — matrix operand order repaired and checked by reading, not compiled, not run on hardware
//
#include <hip/hip_runtime.h>
#include <stdint.h>

// ---------- types ----------
typedef __attribute__((ext_vector_type(16))) __bf16 v16bf;
typedef __attribute__((ext_vector_type(8)))  float  v8f;

union Frag {
  v16bf v;
  unsigned short s[16];
  uint4 q[2];
};

__device__ __forceinline__ unsigned short f32_bf16(float f) {
  union { __bf16 h; unsigned short u; } c;
  c.h = (__bf16)f;                          // RNE convert -> v_cvt bf16
  return c.u;
}

__device__ __forceinline__ v8f vzero8() {
  v8f z;
#pragma unroll
  for (int i = 0; i < 8; ++i) z[i] = 0.0f;
  return z;
}

// =====================================================================
// GEMM:  Out[M,N] = A[M,K] * W[K,N] + bias[N]
// A fp32 or bf16 (template), W/bias fp32, Out fp32 or bf16 (template).
// Block tile 128x128x32, 256 threads (8 waves), wave tile 64x32.
// Register-pipelined: global loads for slice k+1 issue before slice k's
// WMMAs so HBM latency overlaps compute.
// =====================================================================
template <bool A_BF16, bool OUT_BF16>
__global__ __launch_bounds__(256)
void gemm_bias_wmma(const void* __restrict__ Ap, const float* __restrict__ Bw,
                    const float* __restrict__ bias, void* __restrict__ Outp,
                    int M, int N, int K) {
  constexpr int BM = 128, BN = 128, BK = 32;
  constexpr int LDA = BK + 8;                 // 40 elems -> 80B row (16B aligned)
  __shared__ unsigned short Al[BM * LDA];     // [m][k] bf16
  __shared__ unsigned short Bl[BN * LDA];     // [n][k] bf16 (transposed W tile)

  const int tid  = threadIdx.x;
  const int lane = tid & 31;
  const int w    = tid >> 5;
  const int wm   = w >> 2;                    // 0..1 (64-row block)
  const int wn   = w & 3;                     // 0..3 (32-col block)
  const int mblk = blockIdx.y * BM;
  const int nblk = blockIdx.x * BN;

  const int l15   = lane & 15;
  const int hi    = lane >> 4;                // 0 or 1
  const int koffs = hi << 3;                  // 0 or 8  (A frag)
  const int kb16  = hi << 4;                  // 0 or 16 (unused)

  v8f acc[4][2];
#pragma unroll
  for (int i = 0; i < 4; ++i)
#pragma unroll
    for (int j = 0; j < 2; ++j) acc[i][j] = vzero8();

  // register staging for the global->LDS pipeline
  float4 raf[4];                              // A, fp32 path
  uint4  rab[2];                              // A, bf16 path
  float4 rw[4];                               // W

  // per-thread tile coordinates (fixed)
  const int amA[2] = { (tid + 0)   >> 2, (tid + 256) >> 2 };
  const int akA[2] = { ((tid + 0) & 3) * 8, ((tid + 256) & 3) * 8 };

  auto load_global = [&](int k0) {
#pragma unroll
    for (int p = 0; p < 2; ++p) {
      if (A_BF16) {
        rab[p] = *(const uint4*)((const unsigned short*)Ap +
                   (size_t)(mblk + amA[p]) * K + k0 + akA[p]);
      } else {
        const float* src = (const float*)Ap +
                           (size_t)(mblk + amA[p]) * K + k0 + akA[p];
        raf[2 * p + 0] = *(const float4*)(src);
        raf[2 * p + 1] = *(const float4*)(src + 4);
      }
    }
#pragma unroll
    for (int p = 0; p < 4; ++p) {
      int g  = tid + p * 256;
      int kk = g >> 5;
      int n4 = (g & 31) * 4;
      rw[p] = *(const float4*)(Bw + (size_t)(k0 + kk) * N + nblk + n4);
    }
  };

  auto store_lds = [&]() {
#pragma unroll
    for (int p = 0; p < 2; ++p) {
      if (A_BF16) {
        *(uint4*)&Al[amA[p] * LDA + akA[p]] = rab[p];
      } else {
        union { unsigned short t[8]; uint4 q; } pk;
        pk.t[0] = f32_bf16(raf[2 * p].x);     pk.t[1] = f32_bf16(raf[2 * p].y);
        pk.t[2] = f32_bf16(raf[2 * p].z);     pk.t[3] = f32_bf16(raf[2 * p].w);
        pk.t[4] = f32_bf16(raf[2 * p + 1].x); pk.t[5] = f32_bf16(raf[2 * p + 1].y);
        pk.t[6] = f32_bf16(raf[2 * p + 1].z); pk.t[7] = f32_bf16(raf[2 * p + 1].w);
        *(uint4*)&Al[amA[p] * LDA + akA[p]] = pk.q;
      }
    }
#pragma unroll
    for (int p = 0; p < 4; ++p) {
      int g  = tid + p * 256;
      int kk = g >> 5;
      int n4 = (g & 31) * 4;
      Bl[(n4 + 0) * LDA + kk] = f32_bf16(rw[p].x);
      Bl[(n4 + 1) * LDA + kk] = f32_bf16(rw[p].y);
      Bl[(n4 + 2) * LDA + kk] = f32_bf16(rw[p].z);
      Bl[(n4 + 3) * LDA + kk] = f32_bf16(rw[p].w);
    }
  };

  load_global(0);
  for (int k0 = 0; k0 < K; k0 += BK) {
    store_lds();
    __syncthreads();

    // issue next slice's global loads now; they retire while we WMMA
    if (k0 + BK < K) load_global(k0 + BK);
    // prefetch two slices ahead into GL2
    if (k0 + 2 * BK < K) {
      if (A_BF16)
        __builtin_prefetch((const unsigned short*)Ap +
                           (size_t)(mblk + amA[0]) * K + k0 + 2 * BK, 0, 1);
      else
        __builtin_prefetch((const float*)Ap +
                           (size_t)(mblk + amA[0]) * K + k0 + 2 * BK, 0, 1);
      __builtin_prefetch(Bw + (size_t)(k0 + 2 * BK + (tid >> 5)) * N + nblk, 0, 1);
    }

    // ---- fragments + 8 WMMAs ----
    Frag a[4], bf[2];
#pragma unroll
    for (int i = 0; i < 4; ++i) {
      int m = wm * 64 + i * 16 + l15;
      a[i].q[0] = *(const uint4*)&Al[m * LDA + koffs];
      a[i].q[1] = *(const uint4*)&Al[m * LDA + 16 + koffs];
    }
#pragma unroll
    for (int j = 0; j < 2; ++j) {
      int n = wn * 32 + j * 16 + l15;
      bf[j].q[0] = *(const uint4*)&Bl[n * LDA + koffs];
      bf[j].q[1] = *(const uint4*)&Bl[n * LDA + 16 + koffs];
    }
#pragma unroll
    for (int i = 0; i < 4; ++i)
#pragma unroll
      for (int j = 0; j < 2; ++j)
        acc[i][j] = __builtin_amdgcn_wmma_f32_16x16x32_bf16(
            false, a[i].v, false, bf[j].v, (short)0, acc[i][j], false, false);
    __syncthreads();
  }

  // ---- epilogue: bias + store ----
#pragma unroll
  for (int j = 0; j < 2; ++j) {
    int n = nblk + wn * 32 + j * 16 + l15;
    float bn = bias[n];
#pragma unroll
    for (int i = 0; i < 4; ++i) {
#pragma unroll
      for (int r = 0; r < 8; ++r) {
        int m = mblk + wm * 64 + i * 16 + r + (hi << 3);
        float val = acc[i][j][r] + bn;
        if (OUT_BF16)
          ((unsigned short*)Outp)[(size_t)m * N + n] = f32_bf16(val);
        else
          ((float*)Outp)[(size_t)m * N + n] = val;
      }
    }
  }
}

// =====================================================================
// Flash attention: one WG per (b, h, 128-query tile); 8 waves x 16 rows.
// qkv (bf16) layout [B,T,3C]; y (bf16) layout [B,T,C].
// K tile staged via async global->LDS (ASYNCcnt path); V staged through
// VGPRs because it is transposed on store.
// =====================================================================
__global__ __launch_bounds__(256)
void attn_wmma(const unsigned short* __restrict__ qkv,
               unsigned short* __restrict__ y,
               int T, int C, int H) {
  constexpr int LDK = 72;                     // 144B row (16B aligned)
  __shared__ unsigned short Kl[64 * LDK];     // [key][d]
  __shared__ unsigned short Vt[64 * LDK];     // [d][key] (transposed)
  __shared__ unsigned short Pl[8 * 16 * LDK]; // per-wave P tile [16][LDK]

  const int tid  = threadIdx.x;
  const int lane = tid & 31;
  const int w    = tid >> 5;
  const int l15  = lane & 15;
  const int hi   = lane >> 4;

  const int h = blockIdx.y, b = blockIdx.z;
  const int qbase  = blockIdx.x * 128;
  const int C3     = 3 * C;
  const int qrow_w = qbase + w * 16;          // wave's first q row
  const int wave_qmax = qrow_w + 15;

  // ---- load Q A-fragments straight from global (bf16, 2 k-steps) ----
  Frag aq[2];
  {
    const int qrow = qrow_w + l15;
    const size_t rowbase = ((size_t)(b * T + qrow)) * C3 + h * 64;
    const int ko = hi << 3;
#pragma unroll
    for (int s = 0; s < 2; ++s) {
      aq[s].q[0] = *(const uint4*)(qkv + rowbase + s * 32 + ko);
      aq[s].q[1] = *(const uint4*)(qkv + rowbase + s * 32 + 16 + ko);
    }
  }

  v8f o[4];
#pragma unroll
  for (int j = 0; j < 4; ++j) o[j] = vzero8();
  float mrow[8], lrow[8];
#pragma unroll
  for (int r = 0; r < 8; ++r) { mrow[r] = -1e30f; lrow[r] = 0.0f; }

  const float scale = 0.125f;                 // 1/sqrt(64)
  const int ntiles = qbase / 64 + 2;          // cover keys [0, qbase+128)
  unsigned short* Pw = &Pl[w * 16 * LDK];

  // per-thread tile copy coordinates (fixed)
  const int row0 = tid >> 3;                  // p=0: rows 0..31
  const int c8   = (tid & 7) * 8;

  for (int jt = 0; jt < ntiles; ++jt) {
    const int kbase = jt * 64;
    __syncthreads();                          // protect previous tile's use

    // ---- cooperative K/V tile load ----
#pragma unroll
    for (int p = 0; p < 2; ++p) {
      int row = row0 + p * 32;
      size_t base = ((size_t)(b * T + kbase + row)) * C3 + h * 64;
      // K tile: async DMA global -> LDS (bf16, straight copy)
      unsigned lds_dst = (unsigned)(uintptr_t)&Kl[row * LDK + c8];
      unsigned goff    = (unsigned)((base + C + c8) * 2);     // byte offset
      asm volatile("global_load_async_to_lds_b128 %0, %1, %2"
                   :: "v"(lds_dst), "v"(goff), "s"(qkv) : "memory");
      // V tile: load to VGPRs, transpose on LDS store
      uint4 vv = *(const uint4*)(qkv + base + 2 * C + c8);
      const unsigned short* vs = (const unsigned short*)&vv;
#pragma unroll
      for (int i = 0; i < 8; ++i) Vt[(c8 + i) * LDK + row] = vs[i];
    }
    // prefetch next tile's K/V lines into GL2
    if (jt + 1 < ntiles) {
      size_t nb = ((size_t)(b * T + kbase + 64 + row0)) * C3 + h * 64;
      __builtin_prefetch(qkv + nb + C + c8, 0, 1);
      __builtin_prefetch(qkv + nb + 2 * C + c8, 0, 1);
    }
    asm volatile("s_wait_asynccnt 0x0" ::: "memory");
    __syncthreads();

    if (kbase > wave_qmax) continue;          // wave-uniform: tile fully masked

    // ---- S = Q K^T (8 WMMAs) ----
    v8f sf[4];
#pragma unroll
    for (int j = 0; j < 4; ++j) sf[j] = vzero8();
#pragma unroll
    for (int s = 0; s < 2; ++s) {
#pragma unroll
      for (int j = 0; j < 4; ++j) {
        Frag bk;
        int n = j * 16 + l15;
        bk.q[0] = *(const uint4*)&Kl[n * LDK + s * 32 + (hi << 3)];
        bk.q[1] = *(const uint4*)&Kl[n * LDK + s * 32 + 16 + (hi << 3)];
        sf[j] = __builtin_amdgcn_wmma_f32_16x16x32_bf16(
            false, aq[s].v, false, bk.v, (short)0, sf[j], false, false);
      }
    }

    // ---- scale + causal mask (branchless, EXEC stays full) ----
#pragma unroll
    for (int j = 0; j < 4; ++j) {
      int n = kbase + j * 16 + l15;
#pragma unroll
      for (int r = 0; r < 8; ++r) {
        int q = qrow_w + r + (hi << 3);
        float sv = sf[j][r] * scale;
        sf[j][r] = (n > q) ? -1e30f : sv;
      }
    }

    // ---- online softmax stats (cross-lane over 16-lane half) ----
    float mt[8];
#pragma unroll
    for (int r = 0; r < 8; ++r) {
      float v = sf[0][r];
      v = fmaxf(v, sf[1][r]); v = fmaxf(v, sf[2][r]); v = fmaxf(v, sf[3][r]);
      mt[r] = v;
    }
#pragma unroll
    for (int off = 1; off < 16; off <<= 1)
#pragma unroll
      for (int r = 0; r < 8; ++r)
        mt[r] = fmaxf(mt[r], __shfl_xor(mt[r], off, 32));

    float alpha[8];
#pragma unroll
    for (int r = 0; r < 8; ++r) {
      float nm = fmaxf(mrow[r], mt[r]);
      alpha[r] = __expf(mrow[r] - nm);
      mrow[r] = nm;
    }

    float st[8];
#pragma unroll
    for (int r = 0; r < 8; ++r) st[r] = 0.0f;
#pragma unroll
    for (int j = 0; j < 4; ++j)
#pragma unroll
      for (int r = 0; r < 8; ++r) {
        float p = __expf(sf[j][r] - mrow[r]);
        sf[j][r] = p;
        st[r] += p;
      }
#pragma unroll
    for (int off = 1; off < 16; off <<= 1)
#pragma unroll
      for (int r = 0; r < 8; ++r)
        st[r] += __shfl_xor(st[r], off, 32);
#pragma unroll
    for (int r = 0; r < 8; ++r) lrow[r] = lrow[r] * alpha[r] + st[r];

    // ---- rescale O, stage P as bf16 in per-wave LDS (C->A layout) ----
#pragma unroll
    for (int j = 0; j < 4; ++j)
#pragma unroll
      for (int r = 0; r < 8; ++r) o[j][r] *= alpha[r];
#pragma unroll
    for (int j = 0; j < 4; ++j)
#pragma unroll
      for (int r = 0; r < 8; ++r)
        Pw[(r + (hi << 3)) * LDK + j * 16 + l15] = f32_bf16(sf[j][r]);
    asm volatile("s_wait_dscnt 0x0" ::: "memory");  // same-wave DS RAW ordering

    // ---- O += P V (8 WMMAs) ----
#pragma unroll
    for (int s = 0; s < 2; ++s) {
      Frag ap;
      ap.q[0] = *(const uint4*)&Pw[l15 * LDK + s * 32 + (hi << 3)];
      ap.q[1] = *(const uint4*)&Pw[l15 * LDK + s * 32 + 16 + (hi << 3)];
#pragma unroll
      for (int j = 0; j < 4; ++j) {
        Frag bv;
        int n = j * 16 + l15;
        bv.q[0] = *(const uint4*)&Vt[n * LDK + s * 32 + (hi << 3)];
        bv.q[1] = *(const uint4*)&Vt[n * LDK + s * 32 + 16 + (hi << 3)];
        o[j] = __builtin_amdgcn_wmma_f32_16x16x32_bf16(
            false, ap.v, false, bv.v, (short)0, o[j], false, false);
      }
    }
  }

  // ---- epilogue: O /= l, store y bf16 [B,T,C] ----
#pragma unroll
  for (int r = 0; r < 8; ++r) {
    float rcp = 1.0f / lrow[r];
    int qrow = qrow_w + r + (hi << 3);
    size_t base = ((size_t)(b * T + qrow)) * C + h * 64;
#pragma unroll
    for (int j = 0; j < 4; ++j)
      y[base + j * 16 + l15] = f32_bf16(o[j][r] * rcp);
  }
}

// =====================================================================
// launch
// =====================================================================
extern "C" void kernel_launch(void* const* d_in, const int* in_sizes, int n_in,
                              void* d_out, int out_size, void* d_ws, size_t ws_size,
                              hipStream_t stream) {
  (void)in_sizes; (void)n_in; (void)out_size; (void)ws_size;
  const float* x      = (const float*)d_in[0];
  const float* W_attn = (const float*)d_in[1];
  const float* b_attn = (const float*)d_in[2];
  const float* W_proj = (const float*)d_in[3];
  const float* b_proj = (const float*)d_in[4];
  float* out = (float*)d_out;

  const int B = 4, T = 2048, C = 1024, H = 16;
  const int M = B * T;

  unsigned short* qkv = (unsigned short*)d_ws;                 // [M, 3C] bf16
  unsigned short* yws = qkv + (size_t)M * 3 * C;               // [M, C]  bf16

  dim3 blk(256);

  // 1) qkv = x @ W_attn + b_attn      (fp32 in -> bf16 out)
  dim3 g1((3 * C) / 128, M / 128);
  gemm_bias_wmma<false, true><<<g1, blk, 0, stream>>>(x, W_attn, b_attn, qkv,
                                                      M, 3 * C, C);
  // 2) flash attention               (bf16 -> bf16)
  dim3 g2(T / 128, H, B);
  attn_wmma<<<g2, blk, 0, stream>>>(qkv, yws, T, C, H);

  // 3) out = y @ W_proj + b_proj      (bf16 in -> fp32 out)
  dim3 g3(C / 128, M / 128);
  gemm_bias_wmma<true, false><<<g3, blk, 0, stream>>>(yws, W_proj, b_proj, out,
                                                      M, C, C);
}
